// SelfAttnFuse_36842229465316
// MI455X (gfx1250) — matrix operand order repaired and checked by reading, not compiled, not run on hardware
//
#include <hip/hip_runtime.h>
#include <math.h>

// ---------------------------------------------------------------------------
// SelfAttnFuse for MI455X (gfx1250, wave32, WMMA).
//   BS=8, C=256, C2=128, H=W=64, N=4096.
// Phase 1: bf16 projections Q = Wq*a2+bq  -> Qt [b][i][128]   (row = query)
//                           K = Wk*a1+bk  -> Kt [b][j][128]   (row = key)
//                           V = Wv*m +bv  -> Vb [b][c][j]     (natural layout)
// Phase 2: fused flash attention, O = softmax(Q K^T) V^T, fuse = O * motion.
//          K/V blocks double-buffered in LDS via async global->LDS DMA.
// Workspace: 32 MiB of bf16 intermediates in d_ws.
// ---------------------------------------------------------------------------

#define BSZ   8
#define CV    256
#define CQK   128
#define NPIX  4096
#define KB    64     // keys per block iteration

typedef __attribute__((ext_vector_type(16))) __bf16 v16bf;
typedef __attribute__((ext_vector_type(8)))  float  v8f;
typedef __attribute__((ext_vector_type(4)))  int    v4i;

#if __has_builtin(__builtin_amdgcn_global_load_async_to_lds_b128) && \
    __has_builtin(__builtin_amdgcn_s_wait_asynccnt)
#define HAVE_ASYNC_LDS 1
typedef __attribute__((address_space(1))) v4i gas_v4i;   // global
typedef __attribute__((address_space(3))) v4i las_v4i;   // LDS
#else
#define HAVE_ASYNC_LDS 0
#endif

struct bf16x16_bits { int4 lo; int4 hi; };   // 32 bytes

// Load a 16-element bf16 fragment as two 16B chunks (global or LDS pointers;
// lowers to *_load_b128 pairs).
static __device__ __forceinline__ v16bf load_frag(const __bf16* p0, const __bf16* p1) {
    bf16x16_bits t;
    t.lo = *reinterpret_cast<const int4*>(p0);
    t.hi = *reinterpret_cast<const int4*>(p1);
    return __builtin_bit_cast(v16bf, t);
}

// 16B global -> LDS copy: async DMA (ASYNCcnt) if available, else reg bounce.
static __device__ __forceinline__ void copy16_g2l(const __bf16* g, __bf16* l) {
#if HAVE_ASYNC_LDS
    __builtin_amdgcn_global_load_async_to_lds_b128(
        (gas_v4i*)(g), (las_v4i*)(l), 0, 0);
#else
    *reinterpret_cast<int4*>(l) = *reinterpret_cast<const int4*>(g);
#endif
}

static __device__ __forceinline__ void async_join() {
#if HAVE_ASYNC_LDS
    __builtin_amdgcn_s_wait_asynccnt(0);
#endif
}

// ---------------------------------------------------------------------------
// Projection: out[o, j] = sum_c W[o,c] * X[b,c,j] + bias[o], cast to bf16.
// Threads run over pixels j (coalesced X loads); W/bias loads are uniform
// (scalar loads). OPT output channels per thread.
// outIsCN=1 -> out[b][o][j] (V layout); 0 -> out[b][j][o] (Q/K layout).
// ---------------------------------------------------------------------------
template<int CIN, int OPT>
__global__ __launch_bounds__(256)
void proj_kernel(const float* __restrict__ X, const float* __restrict__ W,
                 const float* __restrict__ bias, __bf16* __restrict__ out,
                 int Cout, int outIsCN)
{
    const int j  = blockIdx.x * 256 + threadIdx.x;
    const int o0 = blockIdx.y * OPT;
    const int b  = blockIdx.z;
    const float* Xb = X + (size_t)b * CIN * NPIX;

    float acc[OPT];
#pragma unroll
    for (int t = 0; t < OPT; ++t) acc[t] = bias[o0 + t];

#pragma unroll 4
    for (int c = 0; c < CIN; ++c) {
        float xv = Xb[(size_t)c * NPIX + j];
#pragma unroll
        for (int t = 0; t < OPT; ++t)
            acc[t] = fmaf(W[(size_t)(o0 + t) * CIN + c], xv, acc[t]);
    }
    if (outIsCN) {
#pragma unroll
        for (int t = 0; t < OPT; ++t)
            out[((size_t)b * Cout + o0 + t) * NPIX + j] = (__bf16)acc[t];
    } else {
#pragma unroll
        for (int t = 0; t < OPT; ++t)
            out[((size_t)b * NPIX + j) * Cout + o0 + t] = (__bf16)acc[t];
    }
}

// ---------------------------------------------------------------------------
// Fused flash attention + elementwise fuse.
// Grid: (N/128 = 32 query blocks, BS). 256 threads = 8 waves.
// Each wave: 16 query rows, O accumulator 16 x (16x16 f32 C-tiles) = 16x256.
// K/V key blocks staged in LDS (double-buffered) once per WG, cutting L2
// fan-out traffic 8x vs per-wave global fragment loads.
// ---------------------------------------------------------------------------
#define KSTR 136     // K LDS row stride (bf16 elems): 64 rows x 128 +pad
#define VSTR 72      // V LDS row stride (bf16 elems): 256 rows x 64 +pad

struct FlashSmem {
    union {
        struct {
            __bf16 kbuf[2][KB][KSTR];       // 34,816 B
            __bf16 vbuf[2][CV][VSTR];       // 73,728 B
        } stage;                            // live during key loop
        float obuf[128][258];               // 132,096 B, live after key loop
    } u;
    struct {
        float  sbuf[16][68];                // S tile fp32 (softmax staging)
        __bf16 pbuf[16][72];                // P tile bf16 (A-frag staging)
        float  stats[16];                   // alpha / 1/l broadcast
    } wv[8];
};

static __device__ __forceinline__ void stage_kv(FlashSmem& sm, int buf,
                                                const __bf16* __restrict__ Kb,
                                                const __bf16* __restrict__ Vp,
                                                int j0, int tid)
{
    // K block: 128 rows x 128 B  = 1024 x 16B chunks (4 per thread, coalesced)
#pragma unroll
    for (int it = 0; it < 4; ++it) {
        const int chunk = tid + it * 256;
        const int row   = chunk >> 3;
        const int col   = (chunk & 7) * 8;
        copy16_g2l(Kb + (size_t)(j0 + row) * CQK + col,
                   &sm.u.stage.kbuf[buf][row][col]);
    }
    // V block: 256 rows x 128 B = 2048 x 16B chunks (8 per thread)
#pragma unroll
    for (int it = 0; it < 8; ++it) {
        const int chunk = tid + it * 256;
        const int c     = chunk >> 3;
        const int off   = (chunk & 7) * 8;
        copy16_g2l(Vp + (size_t)c * NPIX + j0 + off,
                   &sm.u.stage.vbuf[buf][c][off]);
    }
}

__global__ __launch_bounds__(256, 1)
void flash_attn_kernel(const __bf16* __restrict__ Qt,   // [BS][N][128]
                       const __bf16* __restrict__ Kt,   // [BS][N][128]
                       const __bf16* __restrict__ Vb,   // [BS][256][N]
                       const float*  __restrict__ motion,
                       float* __restrict__ fuse_out,    // [BS][256][N]
                       float* __restrict__ attn_out)    // [BS][256][N]
{
    __shared__ FlashSmem sm;

    const int b    = blockIdx.y;
    const int qb   = blockIdx.x;
    const int tid  = threadIdx.x;
    const int w    = tid >> 5;
    const int lane = tid & 31;
    const int l16  = lane & 15;
    const int hx   = lane >> 4;       // half-wave index

    const int i0 = qb * 128 + w * 16;

    const __bf16* Qw = Qt + ((size_t)b * NPIX + i0) * CQK;
    const __bf16* Kb = Kt + (size_t)b * NPIX * CQK;
    const __bf16* Vp = Vb + (size_t)b * CV * NPIX;

    float*  Sbuf  = &sm.wv[w].sbuf[0][0];
    __bf16* Pbuf  = &sm.wv[w].pbuf[0][0];
    float*  stats = &sm.wv[w].stats[0];

    // --- Q A-fragments (16x32 bf16 each), resident for the whole kernel ---
    v16bf Qa[4];
#pragma unroll
    for (int ks = 0; ks < 4; ++ks) {
        const __bf16* p = Qw + (size_t)l16 * CQK + ks * 32 + hx * 8;
        Qa[ks] = load_frag(p, p + 16);
    }

    const v8f vzero = {0.f,0.f,0.f,0.f,0.f,0.f,0.f,0.f};
    v8f O[16];
#pragma unroll
    for (int t = 0; t < 16; ++t) O[t] = vzero;

    float mrow = -INFINITY;   // running row max   (row = l16, both halves agree)
    float lrow = 0.f;         // running row sumexp

    // Prime buffer 0.
    stage_kv(sm, 0, Kb, Vp, 0, tid);
    async_join();
    __syncthreads();

    int cur = 0;
    for (int j0 = 0; j0 < NPIX; j0 += KB) {
        // Kick off async staging of the next key block into the other buffer.
        if (j0 + KB < NPIX)
            stage_kv(sm, cur ^ 1, Kb, Vp, j0 + KB, tid);
        // L2 prefetch two blocks ahead.
        if (j0 + 2 * KB < NPIX)
            __builtin_prefetch(Kb + (size_t)(j0 + 2 * KB + lane) * CQK, 0, 1);

        // ---- S = Q * K^T : 4 j-tiles x 4 k-steps of 16x16x32 bf16 WMMA ----
        v8f S[4];
#pragma unroll
        for (int jt = 0; jt < 4; ++jt) {
            v8f acc = vzero;
#pragma unroll
            for (int ks = 0; ks < 4; ++ks) {
                const __bf16* kp = &sm.u.stage.kbuf[cur][jt * 16 + l16][ks * 32 + hx * 8];
                v16bf Kf = load_frag(kp, kp + 16);
                acc = __builtin_amdgcn_wmma_f32_16x16x32_bf16(
                        false, Qa[ks], false, Kf, (short)0, acc, false, false);
            }
            S[jt] = acc;
        }

        // ---- stage S to per-wave LDS (C/D layout -> row-major) ----
#pragma unroll
        for (int jt = 0; jt < 4; ++jt)
#pragma unroll
            for (int r = 0; r < 8; ++r)
                Sbuf[(r + 8 * hx) * 68 + jt * 16 + l16] = S[jt][r];
        __builtin_amdgcn_wave_barrier();

        // ---- online softmax: lane handles row l16, columns [hx*32, +32) ----
        float sv[32];
        float vmax = -INFINITY;
#pragma unroll
        for (int t = 0; t < 32; ++t) {
            sv[t] = Sbuf[l16 * 68 + hx * 32 + t];
            vmax = fmaxf(vmax, sv[t]);
        }
        vmax = fmaxf(vmax, __shfl_xor(vmax, 16, 32));
        const float mnew  = fmaxf(mrow, vmax);
        const float alpha = __expf(mrow - mnew);
        float lsum = 0.f;
#pragma unroll
        for (int t = 0; t < 32; ++t) {
            float p = __expf(sv[t] - mnew);
            lsum += p;
            Pbuf[l16 * 72 + hx * 32 + t] = (__bf16)p;
        }
        lsum += __shfl_xor(lsum, 16, 32);
        lrow = lrow * alpha + lsum;
        mrow = mnew;

        if (hx == 0) stats[l16] = alpha;
        __builtin_amdgcn_wave_barrier();

        // ---- rescale O accumulators by per-row alpha ----
        float ar[8];
#pragma unroll
        for (int r = 0; r < 8; ++r) ar[r] = stats[r + 8 * hx];
#pragma unroll
        for (int ct = 0; ct < 16; ++ct)
#pragma unroll
            for (int r = 0; r < 8; ++r) O[ct][r] *= ar[r];

        // ---- O += P * V : 2 k-steps x 16 channel tiles ----
#pragma unroll
        for (int ks2 = 0; ks2 < 2; ++ks2) {
            const __bf16* pp = Pbuf + l16 * 72 + ks2 * 32 + hx * 8;
            v16bf Pa = load_frag(pp, pp + 16);
#pragma unroll
            for (int ct = 0; ct < 16; ++ct) {
                const __bf16* vp = &sm.u.stage.vbuf[cur][ct * 16 + l16][ks2 * 32 + hx * 8];
                v16bf Vf = load_frag(vp, vp + 16);
                O[ct] = __builtin_amdgcn_wmma_f32_16x16x32_bf16(
                          false, Pa, false, Vf, (short)0, O[ct], false, false);
            }
        }

        // Next buffer fully landed + everyone done reading current buffer.
        async_join();
        __syncthreads();
        cur ^= 1;
    }

    // ---- normalize by 1/l ----
    const float linv = 1.f / lrow;
    if (hx == 0) stats[l16] = linv;
    __builtin_amdgcn_wave_barrier();
    float lr[8];
#pragma unroll
    for (int r = 0; r < 8; ++r) lr[r] = stats[r + 8 * hx];
#pragma unroll
    for (int ct = 0; ct < 16; ++ct)
#pragma unroll
        for (int r = 0; r < 8; ++r) O[ct][r] *= lr[r];

    // ---- stage O to LDS (overlaps dead K/V buffers), coalesced writeout ----
    __syncthreads();
#pragma unroll
    for (int ct = 0; ct < 16; ++ct)
#pragma unroll
        for (int r = 0; r < 8; ++r)
            sm.u.obuf[w * 16 + r + 8 * hx][ct * 16 + l16] = O[ct][r];
    __syncthreads();

    for (int e = tid; e < 128 * 256; e += 256) {
        const int ii = e & 127;       // local pixel
        const int cc = e >> 7;        // channel
        const float o = sm.u.obuf[ii][cc];
        const size_t gi = ((size_t)b * CV + cc) * NPIX + qb * 128 + ii;
        const float mo = motion[gi];
        attn_out[gi] = o;
        fuse_out[gi] = o * mo;
    }
}

// ---------------------------------------------------------------------------
extern "C" void kernel_launch(void* const* d_in, const int* in_sizes, int n_in,
                              void* d_out, int out_size, void* d_ws, size_t ws_size,
                              hipStream_t stream) {
    (void)in_sizes; (void)n_in; (void)out_size; (void)ws_size;

    const float* a1 = (const float*)d_in[0];
    const float* a2 = (const float*)d_in[1];
    const float* mo = (const float*)d_in[2];
    const float* Wq = (const float*)d_in[3];
    const float* bq = (const float*)d_in[4];
    const float* Wk = (const float*)d_in[5];
    const float* bk = (const float*)d_in[6];
    const float* Wv = (const float*)d_in[7];
    const float* bv = (const float*)d_in[8];

    // Workspace: Qt (8 MiB) | Kt (8 MiB) | Vb (16 MiB)  -- bf16
    __bf16* Qt = (__bf16*)d_ws;
    __bf16* Kt = Qt + (size_t)BSZ * NPIX * CQK;
    __bf16* Vt = Kt + (size_t)BSZ * NPIX * CQK;

    float* fuse = (float*)d_out;
    float* attn = fuse + (size_t)BSZ * CV * NPIX;

    // q = Wq*a2+bq (layout [b][i][o]); k = Wk*a1+bk; v = Wv*m+bv ([b][o][j])
    proj_kernel<CQK, 8><<<dim3(NPIX / 256, CQK / 8, BSZ), 256, 0, stream>>>(a2, Wq, bq, Qt, CQK, 0);
    proj_kernel<CQK, 8><<<dim3(NPIX / 256, CQK / 8, BSZ), 256, 0, stream>>>(a1, Wk, bk, Kt, CQK, 0);
    proj_kernel<CV , 8><<<dim3(NPIX / 256, CV  / 8, BSZ), 256, 0, stream>>>(mo, Wv, bv, Vt, CV, 1);

    flash_attn_kernel<<<dim3(NPIX / 128, BSZ), 256, 0, stream>>>(Qt, Kt, Vt, mo, fuse, attn);
}
